// PyramidNMF_62466004353050
// MI455X (gfx1250) — compile-verified, matrix operand order checked by reading, not run on hardware
//
#include <hip/hip_runtime.h>

typedef __attribute__((ext_vector_type(16))) __bf16 v16bf;
typedef __attribute__((ext_vector_type(8)))  __bf16 v8bf;
typedef __attribute__((ext_vector_type(8)))  float  v8f;
typedef __attribute__((ext_vector_type(4)))  float  v4f;

#define BM 64
#define BN 128
#define BK 32
#define LDSTR 72   // LDS row stride in bf16 (144 B): 16B-aligned, conflict-free b128 reads

// C(MxN) = act(A(MxK) @ B(KxN, or NxK if bT) + bias), f32 in/out, bf16 WMMA compute.
// Requires: M % 64 == 0, N % 128 == 0, K % 32 == 0 (true for all call sites).
// 128 threads = 4 waves (wave32); wave grid 2x2; each wave owns 32x64 (2x4 WMMA tiles).
__global__ __launch_bounds__(128) void gemm_bf16_wmma(
    const float* __restrict__ A, const float* __restrict__ B,
    const float* __restrict__ bias, float* __restrict__ C,
    int M, int N, int K, int relu, int bT)
{
    __shared__ __bf16 As [BM * LDSTR];   // [m][k]
    __shared__ __bf16 BsT[BN * LDSTR];   // [n][k]  (transposed so frag reads are contiguous)

    const int tid  = threadIdx.x;
    const int wave = tid >> 5;
    const int lane = tid & 31;
    const int half = lane >> 4;          // 0: lanes 0-15, 1: lanes 16-31
    const int l16  = lane & 15;
    const int wm   = (wave >> 1) * 32;   // wave row offset within 64
    const int wn   = (wave & 1) * 64;    // wave col offset within 128
    const int bm   = blockIdx.y * BM;
    const int bn   = blockIdx.x * BN;

    // staging: each thread owns 16 contiguous k of one A row and two B rows
    const int sr  = tid & 63;            // 0..63
    const int skh = (tid >> 6) * 16;     // 0 or 16

    v8f acc[2][4] = {};

    for (int k0 = 0; k0 < K; k0 += BK) {
        // ---- stage A: A[bm+sr][k0+skh .. +15], contiguous -> 4x float4 loads
        {
            const v4f* src = (const v4f*)(A + (size_t)(bm + sr) * K + k0 + skh);
            v4f x0 = src[0], x1 = src[1], x2 = src[2], x3 = src[3];
            v8bf p0, p1;
            #pragma unroll
            for (int e = 0; e < 4; ++e) {
                p0[e]     = (__bf16)x0[e];
                p0[e + 4] = (__bf16)x1[e];
                p1[e]     = (__bf16)x2[e];
                p1[e + 4] = (__bf16)x3[e];
            }
            *(v8bf*)&As[sr * LDSTR + skh]     = p0;
            *(v8bf*)&As[sr * LDSTR + skh + 8] = p1;
        }
        // ---- stage B transposed: BsT[n][k] for n = sr and sr+64
        #pragma unroll
        for (int g = 0; g < 2; ++g) {
            const int nn = sr + g * 64;
            v8bf p0, p1;
            if (bT) {
                // B is (N x K) row-major: contiguous in k -> 4x float4 loads
                const v4f* src = (const v4f*)(B + (size_t)(bn + nn) * K + k0 + skh);
                v4f x0 = src[0], x1 = src[1], x2 = src[2], x3 = src[3];
                #pragma unroll
                for (int e = 0; e < 4; ++e) {
                    p0[e]     = (__bf16)x0[e];
                    p0[e + 4] = (__bf16)x1[e];
                    p1[e]     = (__bf16)x2[e];
                    p1[e + 4] = (__bf16)x3[e];
                }
            } else {
                // B is (K x N) row-major: stride-N gather, lane-coalesced across n
                const float* src = B + (size_t)(k0 + skh) * N + bn + nn;
                #pragma unroll
                for (int e = 0; e < 8; ++e) {
                    p0[e] = (__bf16)src[(size_t)e * N];
                    p1[e] = (__bf16)src[(size_t)(e + 8) * N];
                }
            }
            *(v8bf*)&BsT[nn * LDSTR + skh]     = p0;
            *(v8bf*)&BsT[nn * LDSTR + skh + 8] = p1;
        }
        __syncthreads();

        // ---- WMMA: lane<16 holds K {0..7,16..23}, lane>=16 holds K {8..15,24..31}
        // Keep all 4 B fragments resident; stream 2 A fragments against them.
        v16bf bfrag[4];
        #pragma unroll
        for (int j = 0; j < 4; ++j) {
            const __bf16* bb = &BsT[(wn + j * 16 + l16) * LDSTR + half * 8];
            v8bf blo = *(const v8bf*)bb;
            v8bf bhi = *(const v8bf*)(bb + 16);
            bfrag[j] = __builtin_shufflevector(blo, bhi,
                0, 1, 2, 3, 4, 5, 6, 7, 8, 9, 10, 11, 12, 13, 14, 15);
        }
        #pragma unroll
        for (int i = 0; i < 2; ++i) {
            const __bf16* ab = &As[(wm + i * 16 + l16) * LDSTR + half * 8];
            v8bf alo = *(const v8bf*)ab;
            v8bf ahi = *(const v8bf*)(ab + 16);
            v16bf afrag = __builtin_shufflevector(alo, ahi,
                0, 1, 2, 3, 4, 5, 6, 7, 8, 9, 10, 11, 12, 13, 14, 15);
            #pragma unroll
            for (int j = 0; j < 4; ++j) {
                acc[i][j] = __builtin_amdgcn_wmma_f32_16x16x32_bf16(
                    false, afrag, false, bfrag[j], (short)0, acc[i][j], false, false);
            }
        }
        __syncthreads();
    }

    // ---- epilogue: C/D layout: VGPR r -> row (r + 8*half), col = l16 in 16x16 tile
    #pragma unroll
    for (int i = 0; i < 2; ++i) {
        #pragma unroll
        for (int j = 0; j < 4; ++j) {
            const int col = bn + wn + j * 16 + l16;
            const float bv = bias ? bias[col] : 0.f;
            #pragma unroll
            for (int r = 0; r < 8; ++r) {
                const int row = bm + wm + i * 16 + r + (half ? 8 : 0);
                float v = acc[i][j][r] + bv;
                if (relu) v = fmaxf(v, 0.f);
                C[(size_t)row * N + col] = v;
            }
        }
    }
}

// Row-wise L2 normalize X (R x H), in place.
__global__ __launch_bounds__(256) void rownorm_kernel(float* __restrict__ X, int H)
{
    __shared__ float red[256];
    const size_t r = blockIdx.x;
    float s = 0.f;
    for (int c = threadIdx.x; c < H; c += 256) {
        float v = X[r * H + c];
        s += v * v;
    }
    red[threadIdx.x] = s;
    __syncthreads();
    for (int st = 128; st > 0; st >>= 1) {
        if (threadIdx.x < st) red[threadIdx.x] += red[threadIdx.x + st];
        __syncthreads();
    }
    const float inv = rsqrtf(red[0]);
    for (int c = threadIdx.x; c < H; c += 256) X[r * H + c] *= inv;
}

// Per-video Gram matrix: G[v][m][n] = dot(vf[v,m,:], vf[v,n,:]), F = 24.
__global__ __launch_bounds__(256) void gram_kernel(
    const float* __restrict__ VF, float* __restrict__ G, int H)
{
    const int v = blockIdx.x;
    const float* base = VF + (size_t)v * 24 * H;
    for (int p = threadIdx.x; p < 24 * 24; p += 256) {
        int m = p / 24, n = p % 24;
        const float* pm = base + (size_t)m * H;
        const float* pn = base + (size_t)n * H;
        float s = 0.f;
        for (int k = 0; k < H; ++k) s = fmaf(pm[k], pn[k], s);
        G[(size_t)v * 576 + p] = s;
    }
}

// 5-iteration NMF-style attention refinement + fused output reduction.
// S layout: [v][f][t] (VF x T). OUT layout: [t][v]. combine: 0=write, 1=max.
__global__ __launch_bounds__(256) void nmf_iter_kernel(
    const float* __restrict__ S, const float* __restrict__ G,
    float* __restrict__ OUT, int T, int V, int combine)
{
    const int v = blockIdx.y;
    const int t = blockIdx.x * 256 + threadIdx.x;

    __shared__ float Gs[576];
    for (int i = threadIdx.x; i < 576; i += 256) Gs[i] = G[(size_t)v * 576 + i];
    __syncthreads();
    if (t >= T) return;

    float s[24], a[24];
    const float* sp = S + (size_t)v * 24 * T + t;
    #pragma unroll
    for (int f = 0; f < 24; ++f) {
        s[f] = sp[(size_t)f * T];
        a[f] = s[f];
    }

    const float eta = 0.15f / 5.f;
    float norm = 1.f;
    for (int it = 0; it < 5; ++it) {
        float d[24];
        #pragma unroll
        for (int m = 0; m < 24; ++m) {
            float acc = 1e-6f;                       // dnm = a@G + EPS
            #pragma unroll
            for (int n = 0; n < 24; ++n) acc = fmaf(a[n], Gs[n * 24 + m], acc);
            d[m] = acc;
        }
        #pragma unroll
        for (int m = 0; m < 24; ++m) {
            float x = a[m] - eta * (d[m] - s[m]);
            a[m] = (x <= 0.f) ? 1e-6f : x;
        }
        float ns = 0.f;                              // a G a^T
        #pragma unroll
        for (int m = 0; m < 24; ++m) {
            float acc = 0.f;
            #pragma unroll
            for (int n = 0; n < 24; ++n) acc = fmaf(a[n], Gs[n * 24 + m], acc);
            ns = fmaf(acc, a[m], ns);
        }
        norm = ns;
    }

    const float inv = rsqrtf(norm);
    float r = 0.f;
    #pragma unroll
    for (int f = 0; f < 24; ++f) r = fmaf(s[f], a[f], r);
    r *= inv;

    const size_t o = (size_t)t * V + v;
    OUT[o] = combine ? fmaxf(OUT[o], r) : r;
}

static void run_layer(const float* text, const float* video,
                      const float* w1, const float* b1,
                      const float* w2, const float* b2,
                      float* ws, float* out,
                      int T, int V, int F, int D, int H, int combine,
                      hipStream_t stream)
{
    const int VF = V * F;
    // workspace layout (float counts sized for Hmax = 1024)
    float* vh = ws;                        // VF * 1024
    float* vf = vh + (size_t)12288 * 1024;
    float* th = vf + (size_t)12288 * 1024; // T * 1024
    float* tf = th + (size_t)512 * 1024;
    float* S  = vh;                        // alias: vh dead once vf computed
    float* G  = th;                        // alias: th dead once tf computed

    dim3 blk(128);
    // text projection: th = relu(text@w1+b1); tf = th@w2+b2
    gemm_bf16_wmma<<<dim3(H / 128, T / 64), blk, 0, stream>>>(
        text, w1, b1, th, T, H, D, 1, 0);
    gemm_bf16_wmma<<<dim3(H / 128, T / 64), blk, 0, stream>>>(
        th, w2, b2, tf, T, H, H, 0, 0);
    // video projection
    gemm_bf16_wmma<<<dim3(H / 128, VF / 64), blk, 0, stream>>>(
        video, w1, b1, vh, VF, H, D, 1, 0);
    gemm_bf16_wmma<<<dim3(H / 128, VF / 64), blk, 0, stream>>>(
        vh, w2, b2, vf, VF, H, H, 0, 0);
    // L2 normalize
    rownorm_kernel<<<T, 256, 0, stream>>>(tf, H);
    rownorm_kernel<<<VF, 256, 0, stream>>>(vf, H);
    // sims[v][f][t] = vf · tf^T : (VF x H) @ (T x H)^T -> (VF x T)
    gemm_bf16_wmma<<<dim3(T / 128, VF / 64), blk, 0, stream>>>(
        vf, tf, nullptr, S, VF, T, H, 0, 1);
    // per-video Gram
    gram_kernel<<<V, 256, 0, stream>>>(vf, G, H);
    // NMF iterations + fused output / cross-layer max
    nmf_iter_kernel<<<dim3((T + 255) / 256, V), 256, 0, stream>>>(
        S, G, out, T, V, combine);
}

extern "C" void kernel_launch(void* const* d_in, const int* in_sizes, int n_in,
                              void* d_out, int out_size, void* d_ws, size_t ws_size,
                              hipStream_t stream)
{
    const float* text  = (const float*)d_in[0];   // (T,1,D)
    const float* video = (const float*)d_in[1];   // (V,F,D)
    const float* w1_0  = (const float*)d_in[2];
    const float* b1_0  = (const float*)d_in[3];
    const float* w2_0  = (const float*)d_in[4];
    const float* b2_0  = (const float*)d_in[5];
    const float* w1_1  = (const float*)d_in[6];
    const float* b1_1  = (const float*)d_in[7];
    const float* w2_1  = (const float*)d_in[8];
    const float* b2_1  = (const float*)d_in[9];

    const int D  = 512;
    const int F  = 24;
    const int T  = in_sizes[0] / D;               // 512
    const int VF = in_sizes[1] / D;               // 12288
    const int V  = VF / F;                        // 512
    const int H0 = in_sizes[3];                   // 1024
    const int H1 = in_sizes[7];                   // 512

    float* ws  = (float*)d_ws;
    float* out = (float*)d_out;

    run_layer(text, video, w1_0, b1_0, w2_0, b2_0, ws, out, T, V, F, D, H0, /*combine=*/0, stream);
    run_layer(text, video, w1_1, b1_1, w2_1, b2_1, ws, out, T, V, F, D, H1, /*combine=*/1, stream);
    (void)n_in; (void)out_size; (void)ws_size;
}
